// RNNDecoder_85993835200943
// MI455X (gfx1250) — matrix operand order repaired and checked by reading, not compiled, not run on hardware
//
#include <hip/hip_runtime.h>
#include <hip/hip_bf16.h>
#include <math.h>

#define V_DIM  32000
#define E_DIM  512
#define H_DIM  1024
#define BS_DIM 128
#define SL_DIM 256

typedef __attribute__((ext_vector_type(16))) __bf16        v16bf;
typedef __attribute__((ext_vector_type(8)))  float         v8f;
typedef __attribute__((ext_vector_type(8)))  unsigned int  v8u;
typedef __attribute__((ext_vector_type(4)))  unsigned int  v4u;
typedef __attribute__((ext_vector_type(8)))  int           v8i;
typedef __attribute__((ext_vector_type(4)))  int           v4i;

// pack two f32 -> two bf16 (RNE); merge via single v_perm_b32
__device__ __forceinline__ unsigned int f2bf_pk(float lo, float hi) {
    unsigned int ulo = __float_as_uint(lo);
    unsigned int uhi = __float_as_uint(hi);
    ulo += 0x7FFFu + ((ulo >> 16) & 1u);
    uhi += 0x7FFFu + ((uhi >> 16) & 1u);
    return __builtin_amdgcn_perm(uhi, ulo, 0x07060302u);
}

// A fragment (16x32 bf16, MxK) from LDS, contiguous-k layout (not the ISA table order):
// lane = half*16 + m;  VGPR v: k = 16*half + 2*v + {0,1}
__device__ __forceinline__ v16bf frag_a(const unsigned short* buf, int row, int stride, int half) {
    v8u f;
    const unsigned short* p = buf + row * stride;
#pragma unroll
    for (int v = 0; v < 8; ++v) {
        int k0 = 16 * half + 2 * v;
        f[v] = *(const unsigned int*)(p + k0);
    }
    return __builtin_bit_cast(v16bf, f);
}

// B fragment (32x16 bf16, KxN) from LDS: lane = half*16 + n; k = 16*half + 2v + {0,1}
__device__ __forceinline__ v16bf frag_b(const unsigned short* buf, int row, int stride, int half) {
    v8u f;
    const unsigned short* p = buf + row * stride + 16 * half;
#pragma unroll
    for (int v = 0; v < 8; ++v) {
        f[v] = *(const unsigned int*)(p + 2 * v);
    }
    return __builtin_bit_cast(v16bf, f);
}

#define WMMA_BF16(a, b, c) \
    __builtin_amdgcn_wmma_f32_16x16x32_bf16(false, (a), false, (b), (short)0, (c), false, false)

// ---------------------------------------------------------------------------
// TDM: DMA one [128 rows x 64 bf16] tile from a row-major bf16 matrix into
// LDS, hardware-padded to a 72-ushort row stride (32 data dwords + 4 pad
// dwords per row).  D# encoding per CDNA5 ISA.
// ---------------------------------------------------------------------------
__device__ __forceinline__ void tdm_load_b_tile(unsigned int lds_off,
                                                unsigned long long gaddr,
                                                int row_len /*elems, = ldb*/) {
    v4u g0;
    g0[0] = 1u;                                            // count=1 (valid user D#)
    g0[1] = lds_off;                                       // lds_addr (bytes)
    g0[2] = (unsigned int)(gaddr & 0xFFFFFFFFu);           // global_addr[31:0]
    g0[3] = (unsigned int)((gaddr >> 32) & 0x1FFFFFFu)     // global_addr[56:32]
            | (2u << 30);                                  // type=2 ("image")
    v8i g1;
    g1[0] = (int)((1u << 16)      // data_size = 2 bytes
                | (1u << 20)      // pad_enable
                | (4u << 22)      // pad_interval: 32 dwords (=128B row)
                | (3u << 25));    // pad_amount: 4 dwords (=16B pad)
    g1[1] = (int)((row_len & 0xFFFF) << 16);                         // tensor_dim0[15:0]
    g1[2] = (int)(((row_len >> 16) & 0xFFFF) | (1024u << 16));       // dim0[31:16] | dim1[15:0]
    g1[3] = (int)(64u << 16);                                        // tile_dim0 = 64
    g1[4] = 128;                                                     // tile_dim1 = 128
    g1[5] = row_len;                                                 // tensor_dim0_stride[31:0]
    g1[6] = 0;
    g1[7] = 0;
    v4i gz = {0, 0, 0, 0};
#if __has_include(<hip/amd_detail/amd_gfx1250_TDM.h>)
    v8i gz8 = {0, 0, 0, 0, 0, 0, 0, 0};
    __builtin_amdgcn_tensor_load_to_lds(g0, g1, gz, gz, gz8, 0);
#else
    __builtin_amdgcn_tensor_load_to_lds(g0, g1, gz, gz, 0);
#endif
}

// ---------------------------------------------------------------------------
// Variant 1: f32 operands staged+converted through LDS.  BM=128 (one M-block).
// Split-K: blockIdx.z selects K-slice, output goes to slab z (C + z*BM*ldc).
// C[m,n] = sum_k A[m,k]*B[n,k] (+bias[n]).
// ---------------------------------------------------------------------------
__global__ __launch_bounds__(256) void gemm_stage(
    const float* __restrict__ A, int lda,
    const float* __restrict__ B, int ldb,
    float* __restrict__ C, int ldc,
    const float* __restrict__ bias, int K) {
    constexpr int BM = 128, BN = 128, BK = 32, LDSD = 40;
    __shared__ __align__(16) unsigned short As[BM * LDSD];
    __shared__ __align__(16) unsigned short Bs[BN * LDSD];

    const int tid  = threadIdx.x;
    const int wave = tid >> 5;
    const int lane = tid & 31;
    const int half = lane >> 4;
    const int l16  = lane & 15;
    const int n0   = blockIdx.x * BN;
    const int m0   = blockIdx.y * BM;

    // split-K slab
    A += (long)blockIdx.z * K;
    B += (long)blockIdx.z * K;
    C += (long)blockIdx.z * (long)BM * ldc;

    v8f acc[8] = {};
    float4 ra[4], rb[4];

#pragma unroll
    for (int j = 0; j < 4; ++j) {
        int i = (tid + 256 * j) * 4;
        int r = i >> 5, c = i & 31;
        ra[j] = *(const float4*)(A + (long)(m0 + r) * lda + c);
        rb[j] = *(const float4*)(B + (long)(n0 + r) * ldb + c);
    }

    for (int k0 = 0; k0 < K; k0 += BK) {
#pragma unroll
        for (int j = 0; j < 4; ++j) {
            int i = (tid + 256 * j) * 4;
            int r = i >> 5, c = i & 31;
            uint2 pa, pb;
            pa.x = f2bf_pk(ra[j].x, ra[j].y);
            pa.y = f2bf_pk(ra[j].z, ra[j].w);
            pb.x = f2bf_pk(rb[j].x, rb[j].y);
            pb.y = f2bf_pk(rb[j].z, rb[j].w);
            *(uint2*)(As + r * LDSD + c) = pa;
            *(uint2*)(Bs + r * LDSD + c) = pb;
        }
        __syncthreads();

        if (k0 + BK < K) {
#pragma unroll
            for (int j = 0; j < 4; ++j) {
                int i = (tid + 256 * j) * 4;
                int r = i >> 5, c = i & 31;
                ra[j] = *(const float4*)(A + (long)(m0 + r) * lda + (k0 + BK) + c);
                rb[j] = *(const float4*)(B + (long)(n0 + r) * ldb + (k0 + BK) + c);
            }
        }
        // prefetch the HBM-bound B stream two chunks ahead (global_prefetch_b8)
        if (k0 + 2 * BK < K) {
#pragma unroll
            for (int j = 0; j < 4; ++j) {
                int i = (tid + 256 * j) * 4;
                int r = i >> 5, c = i & 31;
                __builtin_prefetch((const void*)(B + (long)(n0 + r) * ldb + (k0 + 2 * BK) + c), 0, 0);
            }
        }

        v16bf a = frag_a(As, wave * 16 + l16, LDSD, half);
#pragma unroll
        for (int s = 0; s < 8; ++s) {
            v16bf b = frag_b(Bs, s * 16 + l16, LDSD, half);
            acc[s] = WMMA_BF16(a, b, acc[s]);
        }
        __syncthreads();
    }

#pragma unroll
    for (int s = 0; s < 8; ++s) {
        int n = n0 + s * 16 + l16;
        float bv = bias ? bias[n] : 0.0f;
#pragma unroll
        for (int v = 0; v < 8; ++v) {
            int m = m0 + wave * 16 + 8 * half + v;  // C layout: VGPR v -> M = v + 8*half
            C[(long)m * ldc + n] = acc[s][v] + bv;
        }
    }
}

// ---------------------------------------------------------------------------
// Variant 2: B pre-converted bf16, tiles DMA'd into double-buffered LDS by
// the Tensor Data Mover (overlapped with WMMA via TENSORcnt).  BK=64.
// EPI==0: plain store (+bias). EPI==1: fused attention epilogue -> partials.
// ---------------------------------------------------------------------------
template <int EPI>
__global__ __launch_bounds__(256) void gemm_bdirect(
    const float* __restrict__ A, int lda,
    const unsigned short* __restrict__ Bbf, int ldb,
    float* __restrict__ C, int ldc,
    const float* __restrict__ bias, int K,
    const float* __restrict__ hproj, const float* __restrict__ attn_b,
    const float* __restrict__ vw, float* __restrict__ partial, int Mtotal) {
    constexpr int BM = 128, BN = 128, BK = 64, LDSD = 72;
    __shared__ __align__(16) unsigned short As[BM * LDSD];
    __shared__ __align__(16) unsigned short Bs[2][BN * LDSD];

    const int tid  = threadIdx.x;
    const int wave = tid >> 5;
    const int lane = tid & 31;
    const int half = lane >> 4;
    const int l16  = lane & 15;
    const int n0   = blockIdx.x * BN;
    const int m0   = blockIdx.y * BM;

    // byte address of this block's B panel (rows n0..n0+127)
    const unsigned long long bbase =
        (unsigned long long)(uintptr_t)Bbf + (unsigned long long)((long)n0 * ldb) * 2ull;
    const unsigned int ldsb0 = (unsigned int)(uintptr_t)(&Bs[0][0]);
    const unsigned int ldsb1 = (unsigned int)(uintptr_t)(&Bs[1][0]);

    v8f acc[8] = {};
    float4 ra[8];

#pragma unroll
    for (int j = 0; j < 8; ++j) {
        int i = (tid + 256 * j) * 4;
        int r = i >> 6, c = i & 63;
        ra[j] = *(const float4*)(A + (long)(m0 + r) * lda + c);
    }

    if (wave == 0) tdm_load_b_tile(ldsb0, bbase, ldb);  // chunk 0 -> buffer 0
    int cur = 0;

    for (int k0 = 0; k0 < K; k0 += BK) {
        // stage A chunk (f32 -> bf16) while the TDM fills Bs[cur]
#pragma unroll
        for (int j = 0; j < 8; ++j) {
            int i = (tid + 256 * j) * 4;
            int r = i >> 6, c = i & 63;
            uint2 pa;
            pa.x = f2bf_pk(ra[j].x, ra[j].y);
            pa.y = f2bf_pk(ra[j].z, ra[j].w);
            *(uint2*)(As + r * LDSD + c) = pa;
        }
        if (wave == 0) __builtin_amdgcn_s_wait_tensorcnt(0);  // Bs[cur] complete
        __syncthreads();                                      // publish As + Bs[cur]

        if (k0 + BK < K) {
#pragma unroll
            for (int j = 0; j < 8; ++j) {
                int i = (tid + 256 * j) * 4;
                int r = i >> 6, c = i & 63;
                ra[j] = *(const float4*)(A + (long)(m0 + r) * lda + (k0 + BK) + c);
            }
            if (wave == 0)
                tdm_load_b_tile(cur ? ldsb0 : ldsb1,
                                bbase + (unsigned long long)(k0 + BK) * 2ull, ldb);
        }

#pragma unroll
        for (int kk = 0; kk < BK; kk += 32) {
            v16bf a = frag_a(As + kk, wave * 16 + l16, LDSD, half);
#pragma unroll
            for (int s = 0; s < 8; ++s) {
                v16bf b = frag_b(Bs[cur] + kk, s * 16 + l16, LDSD, half);
                acc[s] = WMMA_BF16(a, b, acc[s]);
            }
        }
        __syncthreads();
        cur ^= 1;
    }

    if constexpr (EPI == 0) {
#pragma unroll
        for (int s = 0; s < 8; ++s) {
            int n = n0 + s * 16 + l16;
            float bv = bias ? bias[n] : 0.0f;
#pragma unroll
            for (int v = 0; v < 8; ++v) {
                int m = m0 + wave * 16 + 8 * half + v;
                C[(long)m * ldc + n] = acc[s][v] + bv;
            }
        }
    } else {
        // rows m = b*SL + pos; 128-row tile lies within one batch row (128 | 256)
        int bidx = m0 >> 8;
        float rsum[8] = {0, 0, 0, 0, 0, 0, 0, 0};
#pragma unroll
        for (int s = 0; s < 8; ++s) {
            int n = n0 + s * 16 + l16;
            float hb = hproj[bidx * H_DIM + n] + attn_b[n];
            float vv = vw[n];
#pragma unroll
            for (int v = 0; v < 8; ++v) rsum[v] += tanhf(acc[s][v] + hb) * vv;
        }
#pragma unroll
        for (int v = 0; v < 8; ++v) {
            float x = rsum[v];
            x += __shfl_xor(x, 1);
            x += __shfl_xor(x, 2);
            x += __shfl_xor(x, 4);
            x += __shfl_xor(x, 8);
            if (l16 == 0) {
                int m = m0 + wave * 16 + 8 * half + v;
                partial[(long)blockIdx.x * Mtotal + m] = x;  // deterministic, no atomics
            }
        }
    }
}

// f32 -> bf16 bulk conversion (8 elems/thread, b128 in / b128 out)
__global__ void cvt_bf16_kernel(const float* __restrict__ src,
                                unsigned short* __restrict__ dst, int n) {
    int i = (blockIdx.x * 256 + threadIdx.x) * 8;
    if (i >= n) return;
    float4 x0 = *(const float4*)(src + i);
    float4 x1 = *(const float4*)(src + i + 4);
    uint4 o;
    o.x = f2bf_pk(x0.x, x0.y);
    o.y = f2bf_pk(x0.z, x0.w);
    o.z = f2bf_pk(x1.x, x1.y);
    o.w = f2bf_pk(x1.z, x1.w);
    *(uint4*)(dst + i) = o;
}

// embedding gather -> xbuf[:, 0:512] and outcat[:, 3072:3584]
__global__ void embed_kernel(const int* __restrict__ tok, const float* __restrict__ emb,
                             float* __restrict__ xbuf, float* __restrict__ outcat) {
    int b = blockIdx.x, t = threadIdx.x;  // 128 threads x 4 floats
    long row = (long)tok[b] * E_DIM;
    float4 v = *(const float4*)(emb + row + t * 4);
    *(float4*)(xbuf + b * 2560 + t * 4) = v;
    *(float4*)(outcat + b * 3584 + 3072 + t * 4) = v;
}

// reduce 8 partial score chunks, softmax over SL per batch row -> aprob
__global__ void softmax_kernel(const float* __restrict__ partial, float* __restrict__ aprob) {
    __shared__ float red[8];
    int b = blockIdx.x, t = threadIdx.x;  // 256 threads
    int m = b * SL_DIM + t;
    float x = 0.0f;
#pragma unroll
    for (int c = 0; c < 8; ++c) x += partial[(long)c * (BS_DIM * SL_DIM) + m];

    float mx = x;
#pragma unroll
    for (int o = 16; o >= 1; o >>= 1) mx = fmaxf(mx, __shfl_xor(mx, o));
    if ((t & 31) == 0) red[t >> 5] = mx;
    __syncthreads();
    float mm = red[0];
#pragma unroll
    for (int i = 1; i < 8; ++i) mm = fmaxf(mm, red[i]);

    float e = __expf(x - mm);
    float s = e;
#pragma unroll
    for (int o = 16; o >= 1; o >>= 1) s += __shfl_xor(s, o);
    __syncthreads();
    if ((t & 31) == 0) red[t >> 5] = s;
    __syncthreads();
    float ss = 0.0f;
#pragma unroll
    for (int i = 0; i < 8; ++i) ss += red[i];

    aprob[m] = e / ss;
}

// weighted[b,d] = sum_s a[b,s] * eo[b,s,d] -> xbuf[:, 512:2560], outcat[:, 1024:3072]
__global__ void weighted_kernel(const float* __restrict__ a, const float* __restrict__ eo,
                                float* __restrict__ xbuf, float* __restrict__ outcat) {
    __shared__ float as[SL_DIM];
    int b = blockIdx.x, t = threadIdx.x;  // 256 threads x 8 contiguous d
    as[t] = a[b * SL_DIM + t];
    __syncthreads();
    float acc[8] = {};
    const float* eob = eo + (long)b * SL_DIM * (2 * H_DIM) + t * 8;
    for (int s = 0; s < SL_DIM; ++s) {
        float w = as[s];
        const float4* p = (const float4*)(eob + (long)s * (2 * H_DIM));
        float4 x0 = p[0], x1 = p[1];
        acc[0] += w * x0.x; acc[1] += w * x0.y; acc[2] += w * x0.z; acc[3] += w * x0.w;
        acc[4] += w * x1.x; acc[5] += w * x1.y; acc[6] += w * x1.z; acc[7] += w * x1.w;
    }
    float4* xo = (float4*)(xbuf + b * 2560 + 512 + t * 8);
    float4* oo = (float4*)(outcat + b * 3584 + 1024 + t * 8);
    xo[0] = make_float4(acc[0], acc[1], acc[2], acc[3]);
    xo[1] = make_float4(acc[4], acc[5], acc[6], acc[7]);
    oo[0] = make_float4(acc[0], acc[1], acc[2], acc[3]);
    oo[1] = make_float4(acc[4], acc[5], acc[6], acc[7]);
}

__device__ __forceinline__ float sigm(float x) { return 1.0f / (1.0f + __expf(-x)); }

// LSTM pointwise; sums the 6 split-K gate slabs + both biases. Gate order i,f,g,o.
__global__ void lstm_kernel(const float* __restrict__ gp, const float* __restrict__ bih,
                            const float* __restrict__ bhh, const float* __restrict__ c0,
                            float* __restrict__ outcat, float* __restrict__ hout,
                            float* __restrict__ cout) {
    int idx = blockIdx.x * 256 + threadIdx.x;  // 131072 total
    int b = idx >> 10, n = idx & 1023;
    long base = (long)b * 4096;
    float g4[4];
#pragma unroll
    for (int gi = 0; gi < 4; ++gi) {
        int gn = gi * H_DIM + n;
        float s = bih[gn] + bhh[gn];
#pragma unroll
        for (int sl = 0; sl < 6; ++sl) s += gp[(long)sl * (BS_DIM * 4 * H_DIM) + base + gn];
        g4[gi] = s;
    }
    float ig = sigm(g4[0]);
    float fg = sigm(g4[1]);
    float gg = tanhf(g4[2]);
    float og = sigm(g4[3]);
    float c = fg * c0[idx] + ig * gg;
    float h = og * tanhf(c);
    cout[idx] = c;
    hout[idx] = h;
    outcat[(long)b * 3584 + n] = h;
}

extern "C" void kernel_launch(void* const* d_in, const int* in_sizes, int n_in,
                              void* d_out, int out_size, void* d_ws, size_t ws_size,
                              hipStream_t stream) {
    const int*   tok    = (const int*)d_in[0];
    const float* h0     = (const float*)d_in[1];   // [1,128,1024]
    const float* c0     = (const float*)d_in[2];
    const float* eo     = (const float*)d_in[3];   // [128,256,2048]
    const float* emb_w  = (const float*)d_in[4];
    const float* attn_w = (const float*)d_in[5];   // [1024,3072]
    const float* attn_b = (const float*)d_in[6];
    const float* v_w    = (const float*)d_in[7];
    const float* w_ih   = (const float*)d_in[8];   // [4096,2560]
    const float* w_hh   = (const float*)d_in[9];   // [4096,1024]
    const float* b_ih   = (const float*)d_in[10];
    const float* b_hh   = (const float*)d_in[11];
    const float* fco_w  = (const float*)d_in[12];  // [32000,3584]
    const float* fco_b  = (const float*)d_in[13];

    float* ws      = (float*)d_ws;
    float* hproj   = ws;                 //  131072  [128,1024]
    float* partial = hproj + 131072;     //  262144  [8,32768]
    float* aprob   = partial + 262144;   //   32768  [128,256]
    float* xbuf    = aprob + 32768;      //  327680  [128,2560]
    float* gatesp  = xbuf + 327680;      // 3145728  [6][128,4096] split-K slabs
    float* outcat  = gatesp + 3145728;   //  458752  [128,3584]
    unsigned short* awbf = (unsigned short*)(outcat + 458752);  // attn_w bf16 [1024,3072]

    float* pred = (float*)d_out;                 // [128,32000]
    float* hout = pred + (long)BS_DIM * V_DIM;   // [128,1024]
    float* cout = hout + BS_DIM * H_DIM;         // [128,1024]

    // 0) attn_w f32 -> bf16 once (3.1M elems, stays L2-resident)
    cvt_bf16_kernel<<<1536, 256, 0, stream>>>(attn_w, awbf, H_DIM * 3 * H_DIM);

    // 1) embedding gather
    embed_kernel<<<BS_DIM, 128, 0, stream>>>(tok, emb_w, xbuf, outcat);

    // 2) h_proj = h_top @ attn_w[:, :H].T   M=128 N=1024 K=1024  (TDM-fed B)
    gemm_bdirect<0><<<dim3(8, 1), 256, 0, stream>>>(
        h0, H_DIM, awbf, 3 * H_DIM, hproj, H_DIM,
        nullptr, H_DIM, nullptr, nullptr, nullptr, nullptr, 0);

    // 3) fused energies: tanh(eo @ attn_w[:,H:].T + h_proj + b) . v -> partial scores
    //    M=32768 N=1024 K=2048  (TDM-fed B, BK=64)
    gemm_bdirect<1><<<dim3(8, 256), 256, 0, stream>>>(
        eo, 2 * H_DIM, awbf + H_DIM, 3 * H_DIM, nullptr, 0,
        nullptr, 2 * H_DIM, hproj, attn_b, v_w, partial, BS_DIM * SL_DIM);

    // 4) softmax (reduces 8 N-chunk partials)
    softmax_kernel<<<BS_DIM, SL_DIM, 0, stream>>>(partial, aprob);

    // 5) context vector
    weighted_kernel<<<BS_DIM, 256, 0, stream>>>(aprob, eo, xbuf, outcat);

    // 6) gates slabs 0..3: x @ w_ih.T   M=128 N=4096, split-K 4x640 (128 blocks)
    gemm_stage<<<dim3(32, 1, 4), 256, 0, stream>>>(
        xbuf, 2560, w_ih, 2560, gatesp, 4096, nullptr, 640);

    // 7) gates slabs 4..5: h_top @ w_hh.T   split-K 2x512 (64 blocks)
    gemm_stage<<<dim3(32, 1, 2), 256, 0, stream>>>(
        h0, H_DIM, w_hh, H_DIM, gatesp + 4L * BS_DIM * 4 * H_DIM, 4096, nullptr, 512);

    // 8) LSTM pointwise (sums slabs + biases) -> h_new, c_new (also outcat[:, :1024])
    lstm_kernel<<<512, 256, 0, stream>>>(gatesp, b_ih, b_hh, c0, outcat, hout, cout);

    // 9) prediction = outcat @ fco_w.T + fco_b   M=128 N=32000 K=3584 (250 blocks)
    gemm_stage<<<dim3(250, 1, 1), 256, 0, stream>>>(
        outcat, 3584, fco_w, 3584, pred, V_DIM, fco_b, 3584);
}
